// up_block_3040836845606
// MI455X (gfx1250) — matrix operand order repaired and checked by reading, not compiled, not run on hardware
//
#include <hip/hip_runtime.h>
#include <hip/hip_bf16.h>

typedef __attribute__((ext_vector_type(16))) _Float16 v16h;
typedef __attribute__((ext_vector_type(8)))  _Float16 v8h;
typedef __attribute__((ext_vector_type(8)))  float    v8f;

#define B_    8
#define C_    128
#define N_    1024
#define M_    4096      // up * N
#define CIN   130
#define CQ    32
#define HPAD  144       // 9 * 16   (h rows padded)
#define XPAD  160       // 5 * 32   (x rows padded, K-loop friendly)
#define HP_LDW 40       // LDS row stride (halves) for staged H tile: 80B = 20
                        // dwords -> 16 consecutive rows hit 16 distinct banks

// ---------------------------------------------------------------- WMMA helpers

__device__ __forceinline__ v8f wmma_f16f32(v16h a, v16h b, v8f c) {
  // D = A(16x32 f16) * B(32x16 f16) + C(16x16 f32)
  return __builtin_amdgcn_wmma_f32_16x16x32_f16(false, a, false, b, (short)0, c,
                                                false, false);
}

// A operand (16 rows x K=32, f16) from row-major [rows][ld] buffer.
// ISA layout: lane<16 holds row (lane), halves = K {k0..k0+7, k0+16..k0+23};
//             lane>=16 holds row (lane-16), halves = K {k0+8..15, k0+24..31}.
__device__ __forceinline__ v16h load_a(const _Float16* base, int ld, int r0,
                                       int k0, int lane) {
  int row = r0 + (lane & 15);
  int cb  = k0 + ((lane & 16) ? 8 : 0);
  const _Float16* p = base + (size_t)row * ld + cb;
  v8h lo = *(const v8h*)p;
  v8h hi = *(const v8h*)(p + 16);
  v16h a;
#pragma unroll
  for (int i = 0; i < 8; ++i) { a[i] = lo[i]; a[i + 8] = hi[i]; }
  return a;
}

// Same A-operand gather but from an LDS-resident tile with HP_LDW row stride.
__device__ __forceinline__ v16h load_a_lds(const _Float16* base, int r0,
                                           int lane) {
  int row = r0 + (lane & 15);
  int cb  = (lane & 16) ? 8 : 0;
  const _Float16* p = base + row * HP_LDW + cb;
  v8h lo = *(const v8h*)p;
  v8h hi = *(const v8h*)(p + 16);
  v16h a;
#pragma unroll
  for (int i = 0; i < 8; ++i) { a[i] = lo[i]; a[i + 8] = hi[i]; }
  return a;
}

// B operand (K=32 rows x 16 cols, f16) from row-major [K][ld] buffer.
// ISA layout: lane holds column N=(lane&15), halves = K {k0+8h..+7, k0+16+8h..+7}, h=lane>>4.
__device__ __forceinline__ v16h load_b(const _Float16* base, int ld, int k0,
                                       int n0, int lane) {
  int n  = n0 + (lane & 15);
  int kb = k0 + ((lane & 16) ? 8 : 0);
  v16h b;
#pragma unroll
  for (int i = 0; i < 8; ++i) {
    b[i]     = base[(size_t)(kb + i) * ld + n];
    b[i + 8] = base[(size_t)(kb + 16 + i) * ld + n];
  }
  return b;
}

// ---------------------------------------------------------------- prep kernel
// f16-convert W1 (pad 130->160 K) and W2; zero xp pad rows [144,160).

__global__ void k_prep(const float* __restrict__ W1, const float* __restrict__ W2,
                       _Float16* __restrict__ W1p, _Float16* __restrict__ W2p,
                       _Float16* __restrict__ xp) {
  int idx = blockIdx.x * blockDim.x + threadIdx.x;
  const int n1 = 256 * XPAD;
  const int n2 = 128 * 256;
  const int n3 = B_ * (XPAD - HPAD) * M_;
  if (idx < n1) {
    int r = idx / XPAD, c = idx % XPAD;
    W1p[idx] = (c < CIN) ? (_Float16)W1[r * CIN + c] : (_Float16)0.f;
  } else if ((idx -= n1) < n2) {
    W2p[idx] = (_Float16)W2[idx];
  } else if ((idx -= n2) < n3) {
    int b   = idx / ((XPAD - HPAD) * M_);
    int rem = idx % ((XPAD - HPAD) * M_);
    int c   = HPAD + rem / M_;
    int m   = rem % M_;
    xp[((size_t)b * XPAD + c) * M_ + m] = (_Float16)0.f;
  }
}

// ---------------------------------------------------------------- f / g / h
// F  [B][32][M] f16 (row-major, B-operand friendly)
// Gt [B][M][32] f16 (transposed, A-operand friendly for S = G^T F)
// Hp [B][144][M] f16 (rows 130..143 zero)

__global__ void __launch_bounds__(256) k_fgh(
    const float* __restrict__ inputs,
    const float* __restrict__ WF, const float* __restrict__ bF,
    const float* __restrict__ WG, const float* __restrict__ bG,
    const float* __restrict__ WH, const float* __restrict__ bH,
    _Float16* __restrict__ F, _Float16* __restrict__ Gt,
    _Float16* __restrict__ Hp) {
  __shared__ float net_s[CIN * 64];
  int b  = blockIdx.x / (M_ / 64);
  int m0 = (blockIdx.x % (M_ / 64)) * 64;
  int g  = m0 >> 10;                       // grid row (64 | 1024, never splits)
  float gx = (g < 2) ? -0.2f : 0.2f;
  float gy = (g & 1) ? 0.2f : -0.2f;
  int nbase = m0 & (N_ - 1);
  for (int i = threadIdx.x; i < CIN * 64; i += blockDim.x) {
    int c = i >> 6, m = i & 63;
    float v;
    if (c < C_)      v = inputs[((size_t)b * C_ + c) * N_ + nbase + m];
    else             v = (c == C_) ? gx : gy;
    net_s[i] = v;
  }
  __syncthreads();
  // outputs: 32 (F) + 32 (G) + 144 (Hp incl. zero pad) channels x 64 cols
  for (int i = threadIdx.x; i < (2 * CQ + HPAD) * 64; i += blockDim.x) {
    int co = i >> 6, m = i & 63;
    const float* w; float acc; _Float16* dst;
    if (co < CQ) {
      w = WF + co * CIN; acc = bF[co];
      dst = F + ((size_t)b * CQ + co) * M_ + m0 + m;
    } else if (co < 2 * CQ) {
      int c2 = co - CQ;
      w = WG + c2 * CIN; acc = bG[c2];
      dst = Gt + ((size_t)b * M_ + m0 + m) * CQ + c2;
    } else {
      int c3 = co - 2 * CQ;
      if (c3 >= CIN) {           // zero-pad rows of Hp
        Hp[((size_t)b * HPAD + c3) * M_ + m0 + m] = (_Float16)0.f;
        continue;
      }
      w = WH + c3 * CIN; acc = bH[c3];
      dst = Hp + ((size_t)b * HPAD + c3) * M_ + m0 + m;
    }
#pragma unroll 2
    for (int c = 0; c < CIN; ++c) acc += w[c] * net_s[(c << 6) + m];
    *dst = (_Float16)fmaxf(acc, 0.f);
  }
}

// ---------------------------------------------------------------- row stats
// For each (b, k): max_n s[b,k,n] and 1/sum_n exp(s - max). One wave per
// 64-row k-block (4 resident A tiles -> 4 WMMAs per shared F-tile load),
// two WMMA sweeps over n (max, then sum-of-exp).

#define KT 4

__global__ void __launch_bounds__(256) k_stats(
    const _Float16* __restrict__ F, const _Float16* __restrict__ Gt,
    float2* __restrict__ stats) {
  int wid  = threadIdx.x >> 5;
  int lane = threadIdx.x & 31;
  int gw = blockIdx.x * (blockDim.x >> 5) + wid;
  int b  = gw / (M_ / (16 * KT));
  int k0 = (gw % (M_ / (16 * KT))) * (16 * KT);
  const _Float16* Fb  = F  + (size_t)b * CQ * M_;
  const _Float16* Gtb = Gt + (size_t)b * M_ * CQ;
  v8f zero = {};
  v16h a[KT];
#pragma unroll
  for (int t = 0; t < KT; ++t) a[t] = load_a(Gtb, CQ, k0 + 16 * t, 0, lane);
  float mx[KT][8];
#pragma unroll
  for (int t = 0; t < KT; ++t)
#pragma unroll
    for (int r = 0; r < 8; ++r) mx[t][r] = -3.4e38f;
  for (int n0 = 0; n0 < M_; n0 += 16) {
    v16h bm = load_b(Fb, M_, 0, n0, lane);
#pragma unroll
    for (int t = 0; t < KT; ++t) {
      v8f s = wmma_f16f32(a[t], bm, zero);
#pragma unroll
      for (int r = 0; r < 8; ++r) mx[t][r] = fmaxf(mx[t][r], s[r]);
    }
  }
#pragma unroll
  for (int off = 1; off < 16; off <<= 1)        // reduce within 16-lane halves
#pragma unroll
    for (int t = 0; t < KT; ++t)
#pragma unroll
      for (int r = 0; r < 8; ++r)
        mx[t][r] = fmaxf(mx[t][r], __shfl_xor(mx[t][r], off, 32));
  float sm[KT][8];
#pragma unroll
  for (int t = 0; t < KT; ++t)
#pragma unroll
    for (int r = 0; r < 8; ++r) sm[t][r] = 0.f;
  for (int n0 = 0; n0 < M_; n0 += 16) {
    v16h bm = load_b(Fb, M_, 0, n0, lane);
#pragma unroll
    for (int t = 0; t < KT; ++t) {
      v8f s = wmma_f16f32(a[t], bm, zero);
#pragma unroll
      for (int r = 0; r < 8; ++r) sm[t][r] += __expf(s[r] - mx[t][r]);
    }
  }
#pragma unroll
  for (int off = 1; off < 16; off <<= 1)
#pragma unroll
    for (int t = 0; t < KT; ++t)
#pragma unroll
      for (int r = 0; r < 8; ++r) sm[t][r] += __shfl_xor(sm[t][r], off, 32);
  if ((lane & 15) == 0) {
    int rb = k0 + ((lane & 16) ? 8 : 0);
#pragma unroll
    for (int t = 0; t < KT; ++t)
#pragma unroll
      for (int r = 0; r < 8; ++r)
        stats[(size_t)b * M_ + rb + 16 * t + r] =
            make_float2(mx[t][r], 1.f / sm[t][r]);
  }
}

// ---------------------------------------------------------------- attention
// One wave per 16-wide m-tile, 4 waves/block. The H[0:144][k0:k0+32] slice is
// shared by all 4 waves, so it is staged into LDS once per block with
// double-buffered global_load_async_to_lds_b128 (ASYNCcnt), overlapping the
// next tile's DMA with this tile's 11 WMMAs. Per k-step: recompute two S
// tiles, apply softmax stats, LDS-transpose P to B-operand layout, accumulate
// O += H * P over 9 c-tiles. Epilogue: x = gamma*o + net -> xp f16.

__global__ void __launch_bounds__(128) k_attn(
    const _Float16* __restrict__ F, const _Float16* __restrict__ Gt,
    const _Float16* __restrict__ Hp, const float2* __restrict__ stats,
    const float* __restrict__ inputs, const float* __restrict__ gamma_p,
    _Float16* __restrict__ xp) {
  __shared__ float2 s_stats[M_];                       // 32 KB
  __shared__ __align__(32) _Float16 s_tr[4][512];      // 1 KB transpose buf/wave
  __shared__ __align__(16) _Float16 s_hp[2 * HPAD * HP_LDW];  // 22.5 KB, 2 bufs
  int wid  = threadIdx.x >> 5;
  int lane = threadIdx.x & 31;
  int b  = blockIdx.x >> 6;
  int mg = blockIdx.x & 63;
  int m0 = (mg * 4 + wid) * 16;
  for (int i = threadIdx.x; i < M_; i += blockDim.x)
    s_stats[i] = stats[(size_t)b * M_ + i];
  const _Float16* Fb  = F  + (size_t)b * CQ * M_;
  const _Float16* Gtb = Gt + (size_t)b * M_ * CQ;
  const _Float16* Hb  = Hp + (size_t)b * HPAD * M_;
  uint32_t hp_lds0 = (uint32_t)(uintptr_t)&s_hp[0];    // wave-relative LDS addr

  // Async-stage H[0:144][k0:k0+32] (f16) into LDS buffer `buf`.
  // 144 rows x 64 B, one b128 per element, spread over the 128 threads.
  auto stage_hp = [&](int buf, int k0) {
    const _Float16* gb = Hb + k0;                      // row stride M_ halves
    uint32_t lb = hp_lds0 + (uint32_t)buf * (HPAD * HP_LDW * 2);
    for (int e = (int)threadIdx.x; e < HPAD * 4; e += 128) {
      int row = e >> 2, q = e & 3;
      uint32_t voff = (uint32_t)((row * M_ + q * 8) * 2);
      uint32_t la   = lb + (uint32_t)((row * HP_LDW + q * 8) * 2);
      asm volatile("global_load_async_to_lds_b128 %0, %1, %2"
                   :: "v"(la), "v"(voff), "s"(gb) : "memory");
    }
  };

  stage_hp(0, 0);                                      // prefetch first tile
  __syncthreads();                                     // s_stats ready

  v8f zero = {};
  v16h bf = load_b(Fb, M_, 0, m0, lane);   // F[:, m0..m0+15], fixed per wave
  v8f acc[9];
#pragma unroll
  for (int t = 0; t < 9; ++t) acc[t] = zero;
  _Float16* tr = s_tr[wid];
  int rowoff = (lane & 16) ? 8 : 0;
  int col    = lane & 15;
  int buf = 0;
  for (int k0 = 0; k0 < M_; k0 += 32) {
    v16h a0 = load_a(Gtb, CQ, k0,      0, lane);
    v16h a1 = load_a(Gtb, CQ, k0 + 16, 0, lane);
    v8f s0 = wmma_f16f32(a0, bf, zero);
    v8f s1 = wmma_f16f32(a1, bf, zero);
    // this wave's prior-iter LDS reads done + this tile's async data landed:
    asm volatile("s_wait_dscnt 0x0" ::: "memory");
    asm volatile("s_wait_asynccnt 0x0" ::: "memory");
    __syncthreads();
    if (k0 + 32 < M_) stage_hp(buf ^ 1, k0 + 32);      // overlap next DMA
#pragma unroll
    for (int r = 0; r < 8; ++r) {
      float2 st0 = s_stats[k0 + rowoff + r];
      float2 st1 = s_stats[k0 + 16 + rowoff + r];
      float p0 = __expf(s0[r] - st0.x) * st0.y;        // beta[k, m]
      float p1 = __expf(s1[r] - st1.x) * st1.y;
      int li = (col + 2 * rowoff) * 16 + r;
      tr[li]       = (_Float16)p0;                     // [B lane][k slot]
      tr[li + 8]   = (_Float16)p1;
    }
    asm volatile("s_wait_dscnt 0x0" ::: "memory");     // LDS store->load order
    v16h bp = *(const v16h*)(tr + lane * 16);          // B-operand: col N=lane&15, 16 k slots
    const _Float16* hpb = s_hp + buf * (HPAD * HP_LDW);
#pragma unroll
    for (int t = 0; t < 9; ++t) {
      v16h ah = load_a_lds(hpb, t * 16, lane);         // H[c-tile, k0..k0+31]
      acc[t] = wmma_f16f32(ah, bp, acc[t]);
    }
    buf ^= 1;
  }
  float gamma = gamma_p[0];
  int m = m0 + col;
  int g = m >> 10;
  float gx = (g < 2) ? -0.2f : 0.2f;
  float gy = (g & 1) ? 0.2f : -0.2f;
#pragma unroll
  for (int t = 0; t < 9; ++t)
#pragma unroll
    for (int r = 0; r < 8; ++r) {
      int c = t * 16 + rowoff + r;
      float net;
      if (c < C_)           net = inputs[((size_t)b * C_ + c) * N_ + (m & (N_ - 1))];
      else if (c == C_)     net = gx;
      else if (c == C_ + 1) net = gy;
      else                  net = 0.f;                 // pad rows stay zero
      float x = gamma * acc[t][r] + net;
      xp[((size_t)b * XPAD + c) * M_ + m] = (_Float16)x;
    }
}

// ---------------------------------------------------------------- MLP 1
// y1 = relu(W1 @ x + b1): [256 x 160pad] x [160pad x M], WMMA over K.

__global__ void __launch_bounds__(256) k_mlp1(
    const _Float16* __restrict__ W1p, const float* __restrict__ b1,
    const _Float16* __restrict__ xp, _Float16* __restrict__ y1) {
  int wid = threadIdx.x >> 5, lane = threadIdx.x & 31;
  int gw = blockIdx.x * (blockDim.x >> 5) + wid;
  int ot = gw & 15;
  int mt = (gw >> 4) & (M_ / 16 - 1);
  int b  = gw >> 12;
  int o0 = ot * 16, m0 = mt * 16;
  const _Float16* xb = xp + (size_t)b * XPAD * M_;
  v8f acc = {};
#pragma unroll
  for (int k0 = 0; k0 < XPAD; k0 += 32) {
    v16h a  = load_a(W1p, XPAD, o0, k0, lane);
    v16h bb = load_b(xb, M_, k0, m0, lane);
    acc = wmma_f16f32(a, bb, acc);
  }
  int rowoff = (lane & 16) ? 8 : 0;
  int m = m0 + (lane & 15);
#pragma unroll
  for (int r = 0; r < 8; ++r) {
    int row = o0 + rowoff + r;
    float y = fmaxf(acc[r] + b1[row], 0.f);
    y1[((size_t)b * 256 + row) * M_ + m] = (_Float16)y;
  }
}

// ---------------------------------------------------------------- MLP 2
// out = relu(W2 @ y1 + b2): [128 x 256] x [256 x M] -> f32 output.

__global__ void __launch_bounds__(256) k_mlp2(
    const _Float16* __restrict__ W2p, const float* __restrict__ b2,
    const _Float16* __restrict__ y1, float* __restrict__ out) {
  int wid = threadIdx.x >> 5, lane = threadIdx.x & 31;
  int gw = blockIdx.x * (blockDim.x >> 5) + wid;
  int ot = gw & 7;
  int mt = (gw >> 3) & (M_ / 16 - 1);
  int b  = gw >> 11;
  int o0 = ot * 16, m0 = mt * 16;
  const _Float16* yb = y1 + (size_t)b * 256 * M_;
  v8f acc = {};
#pragma unroll
  for (int k0 = 0; k0 < 256; k0 += 32) {
    v16h a  = load_a(W2p, 256, o0, k0, lane);
    v16h bb = load_b(yb, M_, k0, m0, lane);
    acc = wmma_f16f32(a, bb, acc);
  }
  int rowoff = (lane & 16) ? 8 : 0;
  int m = m0 + (lane & 15);
#pragma unroll
  for (int r = 0; r < 8; ++r) {
    int row = o0 + rowoff + r;
    out[((size_t)b * 128 + row) * M_ + m] = fmaxf(acc[r] + b2[row], 0.f);
  }
}

// ---------------------------------------------------------------- launcher

extern "C" void kernel_launch(void* const* d_in, const int* in_sizes, int n_in,
                              void* d_out, int out_size, void* d_ws, size_t ws_size,
                              hipStream_t stream) {
  const float* inputs = (const float*)d_in[0];
  const float* WF = (const float*)d_in[1];
  const float* bF = (const float*)d_in[2];
  const float* WG = (const float*)d_in[3];
  const float* bG = (const float*)d_in[4];
  const float* WH = (const float*)d_in[5];
  const float* bH = (const float*)d_in[6];
  const float* gamma = (const float*)d_in[7];
  const float* W1 = (const float*)d_in[8];
  const float* b1 = (const float*)d_in[9];
  const float* W2 = (const float*)d_in[10];
  const float* b2 = (const float*)d_in[11];
  float* out = (float*)d_out;

  size_t off = 0;
  auto carve = [&](size_t bytes) -> void* {
    void* p = (char*)d_ws + off;
    off += (bytes + 255) & ~(size_t)255;
    return p;
  };
  _Float16* F    = (_Float16*)carve((size_t)B_ * CQ * M_ * 2);     // 2 MB
  _Float16* Gt   = (_Float16*)carve((size_t)B_ * M_ * CQ * 2);     // 2 MB
  _Float16* Hp   = (_Float16*)carve((size_t)B_ * HPAD * M_ * 2);   // 9.4 MB
  float2*   stats= (float2*)  carve((size_t)B_ * M_ * 8);          // 256 KB
  _Float16* xp   = (_Float16*)carve((size_t)B_ * XPAD * M_ * 2);   // 10.5 MB
  _Float16* y1   = (_Float16*)carve((size_t)B_ * 256 * M_ * 2);    // 16.8 MB
  _Float16* W1p  = (_Float16*)carve((size_t)256 * XPAD * 2);
  _Float16* W2p  = (_Float16*)carve((size_t)128 * 256 * 2);
  (void)ws_size; (void)in_sizes; (void)n_in; (void)out_size;

  {
    int total = 256 * XPAD + 128 * 256 + B_ * (XPAD - HPAD) * M_;
    k_prep<<<(total + 255) / 256, 256, 0, stream>>>(W1, W2, W1p, W2p, xp);
  }
  k_fgh<<<B_ * (M_ / 64), 256, 0, stream>>>(inputs, WF, bF, WG, bG, WH, bH,
                                            F, Gt, Hp);
  k_stats<<<(B_ * (M_ / (16 * KT))) / 8, 256, 0, stream>>>(F, Gt, stats);
  k_attn<<<B_ * (M_ / 16) / 4, 128, 0, stream>>>(F, Gt, Hp, stats, inputs,
                                                 gamma, xp);
  k_mlp1<<<(B_ * (M_ / 16) * 16) / 8, 256, 0, stream>>>(W1p, b1, xp, y1);
  k_mlp2<<<(B_ * (M_ / 16) * 8) / 8, 256, 0, stream>>>(W2p, b2, y1, out);
}
